// SelfAttention_6949257085608
// MI455X (gfx1250) — compile-verified, matrix operand order checked by reading, not run on hardware
//
#include <hip/hip_runtime.h>
#include <hip/hip_bf16.h>
#include <stdint.h>

// ---------------------------------------------------------------------------
// Causal multi-head self-attention for MI455X (gfx1250, wave32, WMMA).
// All matmuls run through v_wmma_f32_16x16x32_bf16 (bf16 operands, f32 acc).
// Pipeline: qkv_gemm -> flash_attn (online softmax) -> out_gemm.
// bf16 tiles stream into LDS via GLOBAL_LOAD_ASYNC_TO_LDS_B128 (ASYNCcnt).
// ---------------------------------------------------------------------------

typedef __attribute__((ext_vector_type(16))) __bf16 v16bf;
typedef __attribute__((ext_vector_type(8)))  __bf16 v8bf;
typedef __attribute__((ext_vector_type(8)))  float  v8f;

#define CDIM 1024
#define TDIM 2048
#define BDIM 4
#define HEADS 16
#define HD 64

// Build a 16-element bf16 A/B fragment from two aligned 8-element LDS/global
// chunks. Per ISA 7.12.2 (16-bit A 16x32): lanes 0-15 hold K {0..7,16..23},
// lanes 16-31 hold K {8..15,24..31}; caller passes the half-dependent bases.
static __device__ inline v16bf make_frag(const __bf16* p0, const __bf16* p1) {
  v8bf a = *(const v8bf*)p0;
  v8bf b = *(const v8bf*)p1;
  return __builtin_shufflevector(a, b, 0, 1, 2, 3, 4, 5, 6, 7,
                                       8, 9, 10, 11, 12, 13, 14, 15);
}

static __device__ inline v8f wmma_bf16(v16bf a, v16bf b, v8f c) {
  return __builtin_amdgcn_wmma_f32_16x16x32_bf16(false, a, false, b,
                                                 (short)0, c, false, false);
}

// Async 16-byte global -> LDS copy (gfx1250 GLOBAL_LOAD_ASYNC_TO_LDS_B128,
// tracked by ASYNCcnt). LDS destination = low 32 bits of the generic shared
// pointer (LDS aperture truncates addr[31:0] to the LDS byte offset).
static __device__ inline void async_copy_b128(void* lds_dst,
                                              const void* gsrc) {
  uint32_t dst = (uint32_t)(uintptr_t)lds_dst;
  asm volatile("global_load_async_to_lds_b128 %0, %1, off"
               :: "v"(dst), "v"(gsrc) : "memory");
}
static __device__ inline void async_wait0() {
  asm volatile("s_wait_asynccnt 0x0" ::: "memory");
}

// ---------------------------------------------------------------------------
// Kernel 1: fused QKV projection.
//   grid = (N/64, M/128, 3), block = 256 (8 waves, 4x2 wave grid of 32x32).
//   Q,K stored [B,H,T,hd] bf16 (contraction dim hd contiguous -> A/B frags).
//   V stored transposed [B,H,hd,T] bf16 (so PV B-fragments load row-major).
// ---------------------------------------------------------------------------
__global__ __launch_bounds__(256) void qkv_gemm(
    const float* __restrict__ X,
    const float* __restrict__ Wq, const float* __restrict__ bq,
    const float* __restrict__ Wk, const float* __restrict__ bk,
    const float* __restrict__ Wv, const float* __restrict__ bv,
    __bf16* __restrict__ Qout, __bf16* __restrict__ Kout,
    __bf16* __restrict__ Vt) {
  const int z = blockIdx.z;
  const float* W    = (z == 0) ? Wq : (z == 1) ? Wk : Wv;
  const float* bias = (z == 0) ? bq : (z == 1) ? bk : bv;

  const int m0 = blockIdx.y * 128;   // over B*T = 8192
  const int n0 = blockIdx.x * 64;    // over C = 1024

  __shared__ __align__(16) __bf16 As[128][40];  // x tile  (row-major, K contig)
  __shared__ __align__(16) __bf16 Bs[64][40];   // W^T tile (n-major, K contig)

  const int tid  = threadIdx.x;
  const int lane = tid & 31, wave = tid >> 5;
  const int wm = wave & 3, wn = wave >> 2;
  const int lr = lane & 15, half = lane >> 4;

  v8f acc[2][2] = {};

  for (int k0 = 0; k0 < CDIM; k0 += 32) {
    // Batch ALL global loads first so they overlap, then convert+store.
    float4 fa[4], fb[2];
    int ar[4], akc[4], bk_[2], bnc[2];
    for (int i = 0; i < 4; ++i) {
      int chunk = tid + i * 256;                 // A: 128x8 float4 chunks
      ar[i] = chunk >> 3;
      akc[i] = (chunk & 7) * 4;
      fa[i] = *(const float4*)&X[(size_t)(m0 + ar[i]) * CDIM + k0 + akc[i]];
    }
    for (int i = 0; i < 2; ++i) {
      int chunk = tid + i * 256;                 // B: 32x16 float4 chunks
      bk_[i] = chunk >> 4;
      bnc[i] = (chunk & 15) * 4;
      fb[i] = *(const float4*)&W[(size_t)(k0 + bk_[i]) * CDIM + n0 + bnc[i]];
    }
    for (int i = 0; i < 4; ++i) {
      __bf16* dst = &As[ar[i]][akc[i]];
      dst[0] = (__bf16)fa[i].x; dst[1] = (__bf16)fa[i].y;
      dst[2] = (__bf16)fa[i].z; dst[3] = (__bf16)fa[i].w;
    }
    for (int i = 0; i < 2; ++i) {                // transpose scatter W -> Bs
      Bs[bnc[i] + 0][bk_[i]] = (__bf16)fb[i].x;
      Bs[bnc[i] + 1][bk_[i]] = (__bf16)fb[i].y;
      Bs[bnc[i] + 2][bk_[i]] = (__bf16)fb[i].z;
      Bs[bnc[i] + 3][bk_[i]] = (__bf16)fb[i].w;
    }
    __syncthreads();

    v16bf a[2], b[2];
    for (int i = 0; i < 2; ++i) {
      int r = wm * 32 + i * 16 + lr;
      a[i] = make_frag(&As[r][half * 8], &As[r][16 + half * 8]);
    }
    for (int j = 0; j < 2; ++j) {
      int n = wn * 32 + j * 16 + lr;
      b[j] = make_frag(&Bs[n][half * 8], &Bs[n][16 + half * 8]);
    }
    for (int i = 0; i < 2; ++i)
      for (int j = 0; j < 2; ++j)
        acc[i][j] = wmma_bf16(a[i], b[j], acc[i][j]);
    __syncthreads();
  }

  // Epilogue: +bias, convert bf16, scatter to Q/K ([B,H,T,hd]) or V^T.
  for (int i = 0; i < 2; ++i)
    for (int j = 0; j < 2; ++j) {
      const int cn = n0 + wn * 32 + j * 16 + lr;
      const float bval = bias[cn];
      const int h = cn >> 6, d = cn & 63;
      for (int r8 = 0; r8 < 8; ++r8) {
        const int rm = m0 + wm * 32 + i * 16 + 8 * half + r8;
        const int bb = rm >> 11;     // / T
        const int t  = rm & (TDIM - 1);
        const __bf16 val = (__bf16)(acc[i][j][r8] + bval);
        if (z == 2) {
          Vt[(((size_t)bb * HEADS + h) * HD + d) * TDIM + t] = val;
        } else {
          __bf16* O = (z == 0) ? Qout : Kout;
          O[(((size_t)bb * HEADS + h) * TDIM + t) * HD + d] = val;
        }
      }
    }
}

// ---------------------------------------------------------------------------
// Kernel 2: causal flash attention.
//   grid = (T/64, H, B), block = 128 (4 waves; each wave owns 16 query rows).
//   K/V tiles (32 keys) streamed into LDS with async b128 copies; online
//   softmax in f32; P re-shaped D-layout -> A-fragment via LDS staging.
// ---------------------------------------------------------------------------
__global__ __launch_bounds__(128) void flash_attn(
    const __bf16* __restrict__ Q, const __bf16* __restrict__ K,
    const __bf16* __restrict__ Vt, __bf16* __restrict__ Y) {
  const int b = blockIdx.z, h = blockIdx.y;
  const int q0 = blockIdx.x * 64;
  const int tid  = threadIdx.x;
  const int lane = tid & 31, wave = tid >> 5;
  const int lr = lane & 15, half = lane >> 4;

  const size_t bh = (size_t)b * HEADS + h;
  const __bf16* Qb = Q  + bh * TDIM * HD;
  const __bf16* Kb = K  + bh * TDIM * HD;
  const __bf16* Vb = Vt + bh * HD * TDIM;

  __shared__ __align__(16) __bf16 Ks[32][72];      // 32 keys x 64 d
  __shared__ __align__(16) __bf16 Vs[64][40];      // 64 d x 32 s (V^T tile)
  __shared__ __align__(16) __bf16 Ps[4][16][40];   // per-wave P staging 16x32

  // Hoist this wave's Q fragments (16 rows x 64 d -> two k=32 A-fragments).
  const __bf16* qp = Qb + (size_t)(q0 + wave * 16 + lr) * HD;
  v16bf qf[2];
  qf[0] = make_frag(qp + half * 8,      qp + 16 + half * 8);
  qf[1] = make_frag(qp + 32 + half * 8, qp + 48 + half * 8);

  float mrow[8], lsum[8];
  v8f oacc[4] = {};
  for (int j = 0; j < 8; ++j) { mrow[j] = -1e30f; lsum[j] = 0.f; }
  const float scale = 0.125f;  // 1/sqrt(64)

  const int send = q0 + 64;    // causal: keys beyond the block's last query
  for (int s0 = 0; s0 < send; s0 += 32) {
    // Async global->LDS tile streaming (no VGPR round-trip, ASYNCcnt).
    for (int i = 0; i < 2; ++i) {
      int chunk = tid + i * 128;                  // K: 32x64 -> 256 chunks
      int s = chunk >> 3, dc = (chunk & 7) * 8;
      async_copy_b128(&Ks[s][dc], &Kb[(size_t)(s0 + s) * HD + dc]);
    }
    for (int i = 0; i < 2; ++i) {
      int chunk = tid + i * 128;                  // V^T: 64x32 -> 256 chunks
      int d = chunk >> 2, sc = (chunk & 3) * 8;
      async_copy_b128(&Vs[d][sc], &Vb[(size_t)d * TDIM + s0 + sc]);
    }
    async_wait0();
    __syncthreads();

    // Wave-uniform causal skip (EXEC stays all-1s for WMMA).
    const bool active = (s0 <= q0 + wave * 16 + 15);
    if (active) {
      // S = Q K^T : two 16x16 column tiles, K-dim 64 = 2 wmma steps each.
      v8f sacc[2] = {};
      for (int ct = 0; ct < 2; ++ct) {
        const int srow = ct * 16 + lr;
        v16bf b0 = make_frag(&Ks[srow][half * 8],      &Ks[srow][16 + half * 8]);
        v16bf b1 = make_frag(&Ks[srow][32 + half * 8], &Ks[srow][48 + half * 8]);
        sacc[ct] = wmma_bf16(qf[0], b0, sacc[ct]);
        sacc[ct] = wmma_bf16(qf[1], b1, sacc[ct]);
      }

      // Causal mask + per-row max (D layout: n = lr + 16*ct, m = j + 8*half).
      float pmax[8];
      for (int j = 0; j < 8; ++j) {
        const int tq = q0 + wave * 16 + 8 * half + j;
        float v0 = sacc[0][j] * scale;
        float v1 = sacc[1][j] * scale;
        if (s0 + lr      > tq) v0 = -1e30f;
        if (s0 + 16 + lr > tq) v1 = -1e30f;
        sacc[0][j] = v0; sacc[1][j] = v1;
        pmax[j] = fmaxf(v0, v1);
      }
      for (int m = 1; m < 16; m <<= 1)
        for (int j = 0; j < 8; ++j)
          pmax[j] = fmaxf(pmax[j], __shfl_xor(pmax[j], m, 32));

      float corr[8];
      for (int j = 0; j < 8; ++j) {
        const float mn = fmaxf(mrow[j], pmax[j]);
        corr[j] = __expf(mrow[j] - mn);
        mrow[j] = mn;
      }
      float psum[8];
      for (int j = 0; j < 8; ++j) {
        const float p0 = __expf(sacc[0][j] - mrow[j]);
        const float p1 = __expf(sacc[1][j] - mrow[j]);
        sacc[0][j] = p0; sacc[1][j] = p1;
        psum[j] = p0 + p1;
      }
      for (int m = 1; m < 16; m <<= 1)
        for (int j = 0; j < 8; ++j)
          psum[j] += __shfl_xor(psum[j], m, 32);
      for (int j = 0; j < 8; ++j) lsum[j] = lsum[j] * corr[j] + psum[j];
      for (int dt = 0; dt < 4; ++dt)
        for (int j = 0; j < 8; ++j) oacc[dt][j] = oacc[dt][j] * corr[j];

      // P: D layout -> row-major LDS staging (16x32 bf16).
      for (int ct = 0; ct < 2; ++ct)
        for (int j = 0; j < 8; ++j)
          Ps[wave][8 * half + j][ct * 16 + lr] = (__bf16)sacc[ct][j];
    }
    __syncthreads();

    if (active) {
      // P @ V : one k=32 wmma per 16-wide d tile.
      v16bf pf = make_frag(&Ps[wave][lr][half * 8],
                           &Ps[wave][lr][16 + half * 8]);
      for (int dt = 0; dt < 4; ++dt) {
        const int dr = dt * 16 + lr;
        v16bf vf = make_frag(&Vs[dr][half * 8], &Vs[dr][16 + half * 8]);
        oacc[dt] = wmma_bf16(pf, vf, oacc[dt]);
      }
    }
    __syncthreads();
  }

  // Finalize: o /= rowsum; store Y as [B,T,C] bf16.
  for (int j = 0; j < 8; ++j) {
    const float inv = 1.0f / lsum[j];
    const int tq = q0 + wave * 16 + 8 * half + j;
    for (int dt = 0; dt < 4; ++dt) {
      const int d = dt * 16 + lr;
      Y[((size_t)b * TDIM + tq) * CDIM + h * HD + d] =
          (__bf16)(oacc[dt][j] * inv);
    }
  }
}

// ---------------------------------------------------------------------------
// Kernel 3: output projection  Out = Y(bf16) @ Wo + bo  -> f32.
//   grid = (N/64, M/128), block = 256. A tile streams via async LDS copies.
// ---------------------------------------------------------------------------
__global__ __launch_bounds__(256) void out_gemm(
    const __bf16* __restrict__ Yb, const float* __restrict__ Wo,
    const float* __restrict__ bo, float* __restrict__ Out) {
  const int m0 = blockIdx.y * 128;
  const int n0 = blockIdx.x * 64;

  __shared__ __align__(16) __bf16 As[128][40];
  __shared__ __align__(16) __bf16 Bs[64][40];

  const int tid  = threadIdx.x;
  const int lane = tid & 31, wave = tid >> 5;
  const int wm = wave & 3, wn = wave >> 2;
  const int lr = lane & 15, half = lane >> 4;

  v8f acc[2][2] = {};

  for (int k0 = 0; k0 < CDIM; k0 += 32) {
    // A tile (bf16 in global): async straight into LDS. 512 chunks, 2/thread.
    for (int i = 0; i < 2; ++i) {
      int chunk = tid + i * 256;
      int r = chunk >> 2;
      int kc = (chunk & 3) * 8;
      async_copy_b128(&As[r][kc], &Yb[(size_t)(m0 + r) * CDIM + k0 + kc]);
    }
    // B tile: Wo[k][n] f32 -> Bs[n][k] bf16 (must convert: VGPR path).
    float4 fb[2];
    int bk_[2], bnc[2];
    for (int i = 0; i < 2; ++i) {
      int chunk = tid + i * 256;
      bk_[i] = chunk >> 4;
      bnc[i] = (chunk & 15) * 4;
      fb[i] = *(const float4*)&Wo[(size_t)(k0 + bk_[i]) * CDIM + n0 + bnc[i]];
    }
    for (int i = 0; i < 2; ++i) {
      Bs[bnc[i] + 0][bk_[i]] = (__bf16)fb[i].x;
      Bs[bnc[i] + 1][bk_[i]] = (__bf16)fb[i].y;
      Bs[bnc[i] + 2][bk_[i]] = (__bf16)fb[i].z;
      Bs[bnc[i] + 3][bk_[i]] = (__bf16)fb[i].w;
    }
    async_wait0();
    __syncthreads();

    v16bf a[2], b[2];
    for (int i = 0; i < 2; ++i) {
      int r = wm * 32 + i * 16 + lr;
      a[i] = make_frag(&As[r][half * 8], &As[r][16 + half * 8]);
    }
    for (int j = 0; j < 2; ++j) {
      int n = wn * 32 + j * 16 + lr;
      b[j] = make_frag(&Bs[n][half * 8], &Bs[n][16 + half * 8]);
    }
    for (int i = 0; i < 2; ++i)
      for (int j = 0; j < 2; ++j)
        acc[i][j] = wmma_bf16(a[i], b[j], acc[i][j]);
    __syncthreads();
  }

  for (int i = 0; i < 2; ++i)
    for (int j = 0; j < 2; ++j) {
      const int cn = n0 + wn * 32 + j * 16 + lr;
      const float bval = bo[cn];
      for (int r8 = 0; r8 < 8; ++r8) {
        const int rm = m0 + wm * 32 + i * 16 + 8 * half + r8;
        Out[(size_t)rm * CDIM + cn] = acc[i][j][r8] + bval;
      }
    }
}

// ---------------------------------------------------------------------------
extern "C" void kernel_launch(void* const* d_in, const int* in_sizes, int n_in,
                              void* d_out, int out_size, void* d_ws,
                              size_t ws_size, hipStream_t stream) {
  const float* x  = (const float*)d_in[0];
  const float* Wq = (const float*)d_in[1];
  const float* bq = (const float*)d_in[2];
  const float* Wk = (const float*)d_in[3];
  const float* bk = (const float*)d_in[4];
  const float* Wv = (const float*)d_in[5];
  const float* bv = (const float*)d_in[6];
  const float* Wo = (const float*)d_in[7];
  const float* bo = (const float*)d_in[8];
  float* out = (float*)d_out;

  // Workspace: Q, K [B,H,T,hd], V^T [B,H,hd,T], Y [B*T,C] — all bf16.
  const size_t NE = (size_t)BDIM * HEADS * TDIM * HD;  // 8,388,608
  __bf16* Q  = (__bf16*)d_ws;
  __bf16* K  = Q + NE;
  __bf16* Vt = K + NE;
  __bf16* Y  = Vt + NE;

  qkv_gemm<<<dim3(CDIM / 64, (BDIM * TDIM) / 128, 3), 256, 0, stream>>>(
      x, Wq, bq, Wk, bk, Wv, bv, Q, K, Vt);
  flash_attn<<<dim3(TDIM / 64, HEADS, BDIM), 128, 0, stream>>>(Q, K, Vt, Y);
  out_gemm<<<dim3(CDIM / 64, (BDIM * TDIM) / 128), 256, 0, stream>>>(
      Y, Wo, bo, out);
}
